// NonLocalBlock_41901700939776
// MI455X (gfx1250) — matrix operand order repaired and checked by reading, not compiled, not run on hardware
//
#include <hip/hip_runtime.h>
#include <hip/hip_bf16.h>

typedef _Float16 half_t;
typedef __attribute__((ext_vector_type(16))) _Float16 v16h;
typedef __attribute__((ext_vector_type(8)))  float    v8f;
typedef int v4i __attribute__((ext_vector_type(4)));

// A-matrix (16x32 f16) per-lane element -> K index (ISA 7.12.2)
__device__ __forceinline__ int a_kidx(int e, int hi) {
    return e + ((e < 8) ? (hi ? 8 : 0) : (hi ? 16 : 8));
}

// 16-byte global -> LDS copy; async-to-LDS on CDNA5 toolchains that expose it.
__device__ __forceinline__ void async_cp16(half_t* lds_dst, const half_t* gsrc) {
#if __has_builtin(__builtin_amdgcn_global_load_async_to_lds_b128)
    __builtin_amdgcn_global_load_async_to_lds_b128(
        (__attribute__((address_space(1))) v4i*)gsrc,
        (__attribute__((address_space(3))) v4i*)lds_dst, 0, 0);
#else
    *(uint4*)lds_dst = *(const uint4*)gsrc;
#endif
}

__device__ __forceinline__ void wait_async() {
#if __has_builtin(__builtin_amdgcn_global_load_async_to_lds_b128)
    asm volatile("s_wait_asynccnt 0" ::: "memory");
#endif
}

// ---------------------------------------------------------------------------
// Kernel 1: fused 1x1-conv projections  Y = x[65536,256] @ [w_theta|w_phi|w_g]
// ---------------------------------------------------------------------------
__global__ __launch_bounds__(128) void proj_kernel(
    const float* __restrict__ x,
    const float* __restrict__ w_theta,
    const float* __restrict__ w_phi,
    const float* __restrict__ w_g,
    half_t* __restrict__ thetaH,   // [65536, 32]
    half_t* __restrict__ phiF,     // [65536, 32]
    half_t* __restrict__ gF)       // [65536, 128]
{
    constexpr int C = 256, NOUT = 192, XP = 264, WP = 264;
    __shared__ half_t xs[64 * XP];         // A tile, padded rows   (33,792 B)
    __shared__ half_t wshT[NOUT * WP];     // weights col-major,pad (101,376 B)

    const int tid  = threadIdx.x;
    const int lane = tid & 31, wv = tid >> 5;
    const int hi   = lane >> 4, l16 = lane & 15;
    const int r0   = blockIdx.x * 64;

    // stage weights transposed: wshT[c * WP + k] = W[k][c]
    for (int i = tid; i < C * NOUT; i += 128) {
        const int k = i / NOUT, c = i % NOUT;
        float v;
        if (c < 32)       v = w_theta[k * 32 + c];
        else if (c < 64)  v = w_phi[k * 32 + (c - 32)];
        else              v = w_g[k * 128 + (c - 64)];
        wshT[c * WP + k] = (half_t)v;
    }
    // stage 64x256 x-tile as f16
    for (int i = tid; i < 64 * C / 4; i += 128) {
        const int row = (i * 4) / C, col = (i * 4) % C;
        const float4 v = ((const float4*)(x + (size_t)r0 * C))[i];
        half_t* d = xs + row * XP + col;
        d[0] = (half_t)v.x; d[1] = (half_t)v.y;
        d[2] = (half_t)v.z; d[3] = (half_t)v.w;
    }
    __syncthreads();

    const half_t* xrow = xs + (wv * 16 + l16) * XP;
    for (int nt = 0; nt < 12; ++nt) {
        const half_t* wcol = wshT + (nt * 16 + l16) * WP;
        v8f acc = {};
#pragma unroll
        for (int k0 = 0; k0 < C; k0 += 32) {
            v16h aa, bb;
#pragma unroll
            for (int e = 0; e < 16; ++e) aa[e] = xrow[k0 + a_kidx(e, hi)];
#pragma unroll
            for (int e = 0; e < 16; ++e) bb[e] = wcol[k0 + a_kidx(e, hi)];
            acc = __builtin_amdgcn_wmma_f32_16x16x32_f16(
                false, aa, false, bb, (short)0, acc, false, false);
        }
        // uniform (scalar) destination select -> no per-lane branch cascade
        half_t* dst; int pitch, cc;
        if (nt < 2)      { dst = thetaH; pitch = 32;  cc = nt * 16 + l16; }
        else if (nt < 4) { dst = phiF;   pitch = 32;  cc = (nt - 2) * 16 + l16; }
        else             { dst = gF;     pitch = 128; cc = (nt - 4) * 16 + l16; }
        const int rowbase = r0 + wv * 16 + (hi ? 8 : 0);
#pragma unroll
        for (int r = 0; r < 8; ++r)
            dst[(size_t)(rowbase + r) * pitch + cc] = (half_t)acc[r];
    }
}

// ---------------------------------------------------------------------------
// Kernel 2: 2x2 max-pool; writes B-fragment-friendly padded/swizzled layouts:
//   phiP : [b*1024 + key][40]            (row = 32 ca + 8 pad halves)
//   gP   : [(b*32+chunk)*128 + cg][40]   (row = 32 keys-in-chunk + 8 pad)
// ---------------------------------------------------------------------------
__global__ __launch_bounds__(256) void pool_kernel(
    const half_t* __restrict__ phiF, const half_t* __restrict__ gF,
    half_t* __restrict__ phiP, half_t* __restrict__ gP)
{
    const int idx = blockIdx.x * 256 + threadIdx.x;
    if (idx >= 16 * 1024 * 160) return;
    const int c  = idx % 160;
    const int bp = idx / 160;
    const int p  = bp & 1023;     // pooled pixel index within batch
    const int b  = bp >> 10;
    const int ph = p >> 5, pw = p & 31;
    const size_t n = (size_t)b * 4096 + (size_t)(ph * 2) * 64 + pw * 2;
    if (c < 32) {
        float v0 = (float)phiF[n * 32 + c];
        float v1 = (float)phiF[(n + 1) * 32 + c];
        float v2 = (float)phiF[(n + 64) * 32 + c];
        float v3 = (float)phiF[(n + 65) * 32 + c];
        phiP[(size_t)bp * 40 + c] = (half_t)fmaxf(fmaxf(v0, v1), fmaxf(v2, v3));
    } else {
        const int cg = c - 32;
        const int chunk = p >> 5;     // 32-key chunk id
        const int kk    = p & 31;     // key within chunk
        float v0 = (float)gF[n * 128 + cg];
        float v1 = (float)gF[(n + 1) * 128 + cg];
        float v2 = (float)gF[(n + 64) * 128 + cg];
        float v3 = (float)gF[(n + 65) * 128 + cg];
        gP[((size_t)(b * 32 + chunk) * 128 + cg) * 40 + kk] =
            (half_t)fmaxf(fmaxf(v0, v1), fmaxf(v2, v3));
    }
}

// ---------------------------------------------------------------------------
// Kernel 3: fused flash attention + output projection + residual
// ---------------------------------------------------------------------------
__global__ __launch_bounds__(128) void attn_kernel(
    const float* __restrict__ x, const float* __restrict__ w_out,
    const float* __restrict__ sigma_p,
    const half_t* __restrict__ thetaH, const half_t* __restrict__ phiP,
    const half_t* __restrict__ gP, float* __restrict__ out)
{
    constexpr int NQ = 4096, NK = 1024, C = 256, CA = 32, CG = 128;
    __shared__ half_t woutT[C * 136];      // w_out col-major, padded (69,632 B)
    __shared__ half_t g_s[CG * 40];        // g chunk  [cg][kk+pad]   (10,240 B)
    __shared__ half_t phi_s[32 * 40];      // phi chunk [key][ca+pad]  (2,560 B)
    __shared__ half_t pstage[4][16 * 40];  // P transpose scratch      (5,120 B)
    __shared__ half_t ostage[4][16 * 136]; // O transpose scratch     (17,408 B)

    const int tid  = threadIdx.x;
    const int lane = tid & 31, wv = tid >> 5;
    const int hi   = lane >> 4, l16 = lane & 15;
    const int b    = blockIdx.y;
    const int q0   = blockIdx.x * 64 + wv * 16;

    // stage w_out transposed: woutT[c * 136 + k] = w_out[k][c]
    for (int i = tid; i < CG * C / 4; i += 128) {
        const int k = (i * 4) / C, c0 = (i * 4) % C;
        const float4 v = ((const float4*)w_out)[i];
        woutT[(c0 + 0) * 136 + k] = (half_t)v.x;
        woutT[(c0 + 1) * 136 + k] = (half_t)v.y;
        woutT[(c0 + 2) * 136 + k] = (half_t)v.z;
        woutT[(c0 + 3) * 136 + k] = (half_t)v.w;
    }

    // theta A-fragment (K = 32 = full head dim); two contiguous 16B runs/lane
    v16h a_th;
    {
        const half_t* tb = thetaH + (size_t)(b * NQ + q0 + l16) * CA;
#pragma unroll
        for (int e = 0; e < 16; ++e) a_th[e] = tb[a_kidx(e, hi)];
    }

    float m_run[8], l_run[8];
    v8f o[8] = {};
#pragma unroll
    for (int r = 0; r < 8; ++r) { m_run[r] = -1e30f; l_run[r] = 0.f; }

    half_t* pst = pstage[wv];
    for (int kc = 0; kc < NK; kc += 32) {
        __syncthreads();
        // async-stage 32-key chunk: phi 1280 halves, g 5120 halves (contiguous)
        {
            const half_t* psrc = phiP + (size_t)(b * NK + kc) * 40;
            for (int i = tid; i < 160; i += 128)
                async_cp16(phi_s + i * 8, psrc + i * 8);
            const half_t* gsrc = gP + (size_t)(b * 32 + (kc >> 5)) * (CG * 40);
            for (int i = tid; i < 640; i += 128)
                async_cp16(g_s + i * 8, gsrc + i * 8);
        }
        wait_async();
        __syncthreads();

        // S = theta @ phi^T : two 16x16 tiles (32 keys)
        v8f s0, s1;
#pragma unroll
        for (int sub = 0; sub < 2; ++sub) {
            v16h bph;
#pragma unroll
            for (int e = 0; e < 16; ++e)
                bph[e] = phi_s[(sub * 16 + l16) * 40 + a_kidx(e, hi)];
            v8f z = {};
            v8f s = __builtin_amdgcn_wmma_f32_16x16x32_f16(
                false, a_th, false, bph, (short)0, z, false, false);
            if (sub == 0) s0 = s; else s1 = s;
        }

        // online softmax: row reductions across each 16-lane half
        float nm[8], scale[8];
#pragma unroll
        for (int r = 0; r < 8; ++r) {
            float mx = fmaxf(s0[r], s1[r]);
#pragma unroll
            for (int msk = 1; msk < 16; msk <<= 1)
                mx = fmaxf(mx, __shfl_xor(mx, msk, 16));
            nm[r] = fmaxf(m_run[r], mx);
            scale[r] = __expf(m_run[r] - nm[r]);
            m_run[r] = nm[r];
        }
#pragma unroll
        for (int r = 0; r < 8; ++r) {
            float p0 = __expf(s0[r] - nm[r]);
            float p1 = __expf(s1[r] - nm[r]);
            float rs = p0 + p1;
#pragma unroll
            for (int msk = 1; msk < 16; msk <<= 1)
                rs += __shfl_xor(rs, msk, 16);
            l_run[r] = l_run[r] * scale[r] + rs;
            const int m = (hi ? 8 : 0) + r;
            pst[m * 40 + l16]      = (half_t)p0;   // transpose P via LDS
            pst[m * 40 + 16 + l16] = (half_t)p1;
        }
        asm volatile("s_wait_dscnt 0" ::: "memory");

#pragma unroll
        for (int t = 0; t < 8; ++t) {
#pragma unroll
            for (int r = 0; r < 8; ++r) o[t][r] *= scale[r];
        }

        // O += P @ g_chunk  (P re-read in A-layout; g fragments contiguous)
        v16h ap;
#pragma unroll
        for (int e = 0; e < 16; ++e) ap[e] = pst[l16 * 40 + a_kidx(e, hi)];
#pragma unroll
        for (int t = 0; t < 8; ++t) {
            v16h bg;
#pragma unroll
            for (int e = 0; e < 16; ++e)
                bg[e] = g_s[(t * 16 + l16) * 40 + a_kidx(e, hi)];
            o[t] = __builtin_amdgcn_wmma_f32_16x16x32_f16(
                false, ap, false, bg, (short)0, o[t], false, false);
        }
    }

    // finalize softmax; transpose attn_g to A-layout via LDS
    float inv[8];
#pragma unroll
    for (int r = 0; r < 8; ++r) inv[r] = 1.f / l_run[r];
    half_t* ost = ostage[wv];
#pragma unroll
    for (int t = 0; t < 8; ++t) {
#pragma unroll
        for (int r = 0; r < 8; ++r) {
            const int m = (hi ? 8 : 0) + r;
            ost[m * 136 + t * 16 + l16] = (half_t)(o[t][r] * inv[r]);
        }
    }
    asm volatile("s_wait_dscnt 0" ::: "memory");

    // out = sigma * (attn_g @ w_out) + x
    const float sg = *sigma_p;
    for (int ct = 0; ct < 16; ++ct) {
        const half_t* wcol = woutT + (ct * 16 + l16) * 136;
        v8f acc = {};
#pragma unroll
        for (int ks = 0; ks < 4; ++ks) {
            v16h aa, bb;
#pragma unroll
            for (int e = 0; e < 16; ++e)
                aa[e] = ost[l16 * 136 + ks * 32 + a_kidx(e, hi)];
#pragma unroll
            for (int e = 0; e < 16; ++e)
                bb[e] = wcol[ks * 32 + a_kidx(e, hi)];
            acc = __builtin_amdgcn_wmma_f32_16x16x32_f16(
                false, aa, false, bb, (short)0, acc, false, false);
        }
#pragma unroll
        for (int r = 0; r < 8; ++r) {
            const int m = (hi ? 8 : 0) + r;
            const size_t idx = (size_t)(b * NQ + q0 + m) * C + ct * 16 + l16;
            out[idx] = sg * acc[r] + x[idx];
        }
    }
}

// ---------------------------------------------------------------------------
extern "C" void kernel_launch(void* const* d_in, const int* in_sizes, int n_in,
                              void* d_out, int out_size, void* d_ws, size_t ws_size,
                              hipStream_t stream) {
    const float* x       = (const float*)d_in[0];
    const float* w_theta = (const float*)d_in[1];
    const float* w_phi   = (const float*)d_in[2];
    const float* w_g     = (const float*)d_in[3];
    const float* w_out   = (const float*)d_in[4];
    const float* sigma   = (const float*)d_in[5];
    float* out = (float*)d_out;

    // workspace carve (f16 elements)
    half_t* ws     = (half_t*)d_ws;
    half_t* thetaH = ws;                                  //  4 MiB
    half_t* phiF   = thetaH + (size_t)16 * 4096 * 32;     //  4 MiB
    half_t* gF     = phiF   + (size_t)16 * 4096 * 32;     // 16 MiB
    half_t* phiP   = gF     + (size_t)16 * 4096 * 128;    // 16*1024*40  (1.25 MiB)
    half_t* gPp    = phiP   + (size_t)16 * 1024 * 40;     // 16*32*128*40 (5 MiB)

    proj_kernel<<<1024, 128, 0, stream>>>(x, w_theta, w_phi, w_g, thetaH, phiF, gF);
    const int total = 16 * 1024 * 160;
    pool_kernel<<<(total + 255) / 256, 256, 0, stream>>>(phiF, gF, phiP, gPp);
    attn_kernel<<<dim3(64, 16), 128, 0, stream>>>(x, w_out, sigma, thetaH, phiP, gPp, out);
}
